// binary_classification_model_16544214024210
// MI455X (gfx1250) — compile-verified, matrix operand order checked by reading, not run on hardware
//
#include <hip/hip_runtime.h>
#include <stdint.h>

// ---------------------------------------------------------------------------
// Model dims: B=32 S=512 E=1024 P=256 K=3 H=8  (M = B*S = 16384)
// GEMM1: 16384x1024 @ 1024x256   (proj)
// GEMM2: 16384x768  @ 768x256    (conv via im2col over shifted proj rows)
// GEMM3: 16384x256  @ 256x4096   (logits) fused with softmax + head-sum in LDS
// All GEMMs: bf16 inputs, f32 accum via v_wmma_f32_16x16x32_bf16 (wave32).
// GEMM3 stages attn_b into LDS with the Tensor Data Mover (TENSORcnt path).
// ---------------------------------------------------------------------------

typedef unsigned short u16;
typedef __attribute__((ext_vector_type(16))) __bf16 v16bf;
typedef __attribute__((ext_vector_type(8)))  float  v8f;
typedef __attribute__((ext_vector_type(4)))  unsigned int v4u;
typedef __attribute__((ext_vector_type(8)))  int v8i_;
typedef __attribute__((ext_vector_type(4)))  int v4i_;

static __device__ __forceinline__ u16 f2bf(float f) {
  union { float f; uint32_t u; } x; x.f = f;
  uint32_t r = 0x7FFFu + ((x.u >> 16) & 1u);
  return (u16)((x.u + r) >> 16);
}
static __device__ __forceinline__ float bf2f(u16 s) {
  union { uint32_t u; float f; } x; x.u = ((uint32_t)s) << 16; return x.f;
}

union FragBF { uint4 q[2]; v16bf v; };

// A fragment (16x32 bf16) from a row-major bf16 matrix with leading dim `ld`.
// Lanes 0-15 hold K {0..7, 16..23}, lanes 16-31 hold K {8..15, 24..31}
// -> two contiguous 16B loads per lane.
static __device__ __forceinline__ v16bf load_a_rm(const u16* A, int row, int ld,
                                                  int kbase, int lane) {
  int half = lane >> 4;
  const u16* p = A + (size_t)row * ld + kbase + half * 8;
  FragBF f;
  f.q[0] = *(const uint4*)(p);
  f.q[1] = *(const uint4*)(p + 16);
  return f.v;
}

// A fragment for the conv GEMM: K-dim column c = tap*256 + in_channel maps to
// proj[row + tap][in_channel]. Runs of 8 never cross a tap boundary.
static __device__ __forceinline__ v16bf load_a_conv(const u16* PR, int row, int kt,
                                                    int lane) {
  int half = lane >> 4;
  int c0 = kt * 32 + half * 8;
  int c1 = c0 + 16;
  int k0 = c0 >> 8, i0 = c0 & 255;
  int k1 = c1 >> 8, i1 = c1 & 255;
  int r0 = row + k0; if (r0 > 16383) r0 = 16383;
  int r1 = row + k1; if (r1 > 16383) r1 = 16383;
  FragBF f;
  f.q[0] = *(const uint4*)(PR + (size_t)r0 * 256 + i0);
  f.q[1] = *(const uint4*)(PR + (size_t)r1 * 256 + i1);
  return f.v;
}

// B fragment (32x16 bf16) from fragment-packed weights: 32 lanes x 16 bf16 contig.
static __device__ __forceinline__ v16bf load_b_pk(const u16* W, int kt, int nt,
                                                  int numNt, int lane) {
  const u16* p = W + ((((size_t)kt * numNt + nt) * 32 + lane) << 4);
  FragBF f;
  f.q[0] = *(const uint4*)(p);
  f.q[1] = *(const uint4*)(p + 8);
  return f.v;
}

static __device__ __forceinline__ v8f wmma_bf16(v16bf a, v16bf b, v8f c) {
  return __builtin_amdgcn_wmma_f32_16x16x32_bf16(false, a, false, b, (short)0, c,
                                                 false, false);
}

// ---------------------------------------------------------------------------
// Prep kernels
// ---------------------------------------------------------------------------
__global__ void cvt_f32_bf16(const float* __restrict__ src, u16* __restrict__ dst,
                             int n) {
  int i = blockIdx.x * blockDim.x + threadIdx.x;
  if (i < n) dst[i] = f2bf(src[i]);
}

// Pack a weight matrix into WMMA B-fragment order.
// mode 0: proj_w  (K=1024, N=256,  src[k*256+n])
// mode 1: conv_w  (K=768,  N=256,  src[(n*256 + (k&255))*3 + (k>>8)])
// mode 2: attn_w  (K=256,  N=4096, src[k*4096+n])
__global__ void pack_w(const float* __restrict__ src, u16* __restrict__ dst,
                       int mode, int total) {
  int i = blockIdx.x * blockDim.x + threadIdx.x;
  if (i >= total) return;
  int j    = i & 15;
  int lane = (i >> 4) & 31;
  int tile = i >> 9;
  int numNt = (mode == 2) ? 256 : 16;
  int nt = tile % numNt;
  int kt = tile / numNt;
  int half = lane >> 4;
  int nin  = lane & 15;
  int kin  = ((j >> 3) << 4) + half * 8 + (j & 7);
  int k = kt * 32 + kin;
  int n = nt * 16 + nin;
  float v;
  if (mode == 0)      v = src[k * 256 + n];
  else if (mode == 1) { int kk = k >> 8, ii = k & 255; v = src[(n * 256 + ii) * 3 + kk]; }
  else                v = src[(size_t)k * 4096 + n];
  dst[i] = f2bf(v);
}

__global__ void zero_f32(float* __restrict__ p, int n) {
  int i = blockIdx.x * blockDim.x + threadIdx.x;
  if (i < n) p[i] = 0.f;
}

// ---------------------------------------------------------------------------
// GEMM1: proj = x @ proj_w + proj_b  -> bf16 (16384 x 256)
// Block = 4 waves, each wave does a 16x64 tile (4 N-subtiles, A reuse x4).
// K loop fully unrolled so the scheduler can overlap loads with WMMA.
// ---------------------------------------------------------------------------
__global__ __launch_bounds__(128) void gemm_proj_kernel(
    const u16* __restrict__ X, const u16* __restrict__ W,
    const float* __restrict__ bias, u16* __restrict__ PR) {
  int lane = threadIdx.x & 31, wave = threadIdx.x >> 5;
  int tm = blockIdx.y * 64 + wave * 16;
  int tn = blockIdx.x * 64;
  int arow = tm + (lane & 15);
  v8f z = {0.f, 0.f, 0.f, 0.f, 0.f, 0.f, 0.f, 0.f};
  v8f acc[4] = {z, z, z, z};
  int nt0 = tn >> 4;
  __builtin_prefetch((const void*)(X + (size_t)arow * 1024), 0, 1);
  #pragma unroll
  for (int kt = 0; kt < 32; ++kt) {
    v16bf a = load_a_rm(X, arow, 1024, kt * 32, lane);
    #pragma unroll
    for (int t = 0; t < 4; ++t) {
      v16bf b = load_b_pk(W, kt, nt0 + t, 16, lane);
      acc[t] = wmma_bf16(a, b, acc[t]);
    }
  }
  int half = lane >> 4, n = lane & 15;
  #pragma unroll
  for (int t = 0; t < 4; ++t) {
    int col = tn + t * 16 + n;
    float bb = bias[col];
    #pragma unroll
    for (int r = 0; r < 8; ++r) {
      int row = tm + half * 8 + r;
      PR[(size_t)row * 256 + col] = f2bf(acc[t][r] + bb);
    }
  }
}

// ---------------------------------------------------------------------------
// GEMM2 (conv): rep_conv[b,s,:] = relu(sum_{i,k} proj[b,s+k,i]*conv_w[o,i,k])
// masked by seq_length; zero for s==0 or s>509.
// ---------------------------------------------------------------------------
__global__ __launch_bounds__(128) void gemm_conv_kernel(
    const u16* __restrict__ PR, const u16* __restrict__ W,
    const float* __restrict__ bias, const int* __restrict__ seqlen,
    u16* __restrict__ RC) {
  int lane = threadIdx.x & 31, wave = threadIdx.x >> 5;
  int tm = blockIdx.y * 64 + wave * 16;
  int tn = blockIdx.x * 64;
  int arow = tm + (lane & 15);
  v8f z = {0.f, 0.f, 0.f, 0.f, 0.f, 0.f, 0.f, 0.f};
  v8f acc[4] = {z, z, z, z};
  int nt0 = tn >> 4;
  #pragma unroll
  for (int kt = 0; kt < 24; ++kt) {
    v16bf a = load_a_conv(PR, arow, kt, lane);
    #pragma unroll
    for (int t = 0; t < 4; ++t) {
      v16bf b = load_b_pk(W, kt, nt0 + t, 16, lane);
      acc[t] = wmma_bf16(a, b, acc[t]);
    }
  }
  int half = lane >> 4, n = lane & 15;
  #pragma unroll
  for (int t = 0; t < 4; ++t) {
    int col = tn + t * 16 + n;
    float bb = bias[col];
    #pragma unroll
    for (int r = 0; r < 8; ++r) {
      int row = tm + half * 8 + r;
      int b_idx = row >> 9;
      int s = row & 511;
      float v = acc[t][r] + bb;
      v = v > 0.f ? v : 0.f;                       // relu
      int l = seqlen[b_idx];
      if (s < 1 || s > 509 || (s - 1) >= (l - 2)) v = 0.f;  // padding + mask
      RC[(size_t)row * 256 + col] = f2bf(v);
    }
  }
}

// ---------------------------------------------------------------------------
// GEMM3 fused: one block per 16-row M-tile; 8 waves cover N=4096 (512 each).
// Logits tile (16x4096 f32 = 256KB) lives in LDS (CDNA5: 320KB/WGP).
// attn_b (16KB) is DMA'd into LDS by the Tensor Data Mover (TENSORcnt).
// Then in-LDS softmax per row + accumulation of sum_{s,h} attn into w[b,q].
// LDS: 256KB logits + 2KB wpart + 16KB bias = 274KB.
// ---------------------------------------------------------------------------
#define LDS_LOGITS_F32  (16 * 4096)
#define LDS_WPART_F32   512
#define LDS_BIAS_OFF_B  ((LDS_LOGITS_F32 + LDS_WPART_F32) * 4)

__global__ __launch_bounds__(256) void attn_softmax_kernel(
    const u16* __restrict__ RC, const u16* __restrict__ W,
    const float* __restrict__ bias, float* __restrict__ wacc) {
  extern __shared__ float smem[];
  float* logits  = smem;                                   // 16 * 4096
  float* wpart   = smem + LDS_LOGITS_F32;                  // 512
  float* biasLds = smem + LDS_LOGITS_F32 + LDS_WPART_F32;  // 4096
  int lane = threadIdx.x & 31, wave = threadIdx.x >> 5;
  int tm = blockIdx.x * 16;
  int b_idx = tm >> 9;                 // 512 rows per batch, tiles aligned

  for (int i = threadIdx.x; i < 512; i += 256) wpart[i] = 0.f;

  // --- TDM: async-copy attn_b (4096 f32) into LDS -------------------------
#if __has_builtin(__builtin_amdgcn_tensor_load_to_lds)
  if (threadIdx.x == 0) {
    unsigned long long ga = (unsigned long long)(uintptr_t)bias;
    // D# group0: count=1 | lds_addr | global_addr[56:0] | type=2
    v4u g0 = { 1u,
               (unsigned)LDS_BIAS_OFF_B,
               (unsigned)(ga & 0xFFFFFFFFu),
               (unsigned)(((ga >> 32) & 0x01FFFFFFu) | (2u << 30)) };
    // D# group1: data_size=4B; tensor 4096x1; tile 4096x1; dim0_stride=4096
    v8i_ g1 = { (int)(2u << 16),     // [17:16] data_size = 4B
                (int)(4096u << 16),  // tensor_dim0[15:0] at bits 63:48
                (int)(1u << 16),     // tensor_dim1 = 1 at bits 111:80
                (int)(4096u << 16),  // tile_dim0 = 4096 at bits 127:112
                1,                   // tile_dim1 = 1
                4096,                // tensor_dim0_stride lo32
                0, 0 };
    v4i_ gz = {0, 0, 0, 0};
#if defined(__clang_major__) && __clang_major__ >= 23
    v8i_ gz8 = {0, 0, 0, 0, 0, 0, 0, 0};
    __builtin_amdgcn_tensor_load_to_lds(g0, g1, gz, gz, gz8, 0);
#else
    __builtin_amdgcn_tensor_load_to_lds(g0, g1, gz, gz, 0);
#endif
  }
  if (wave == 0) __builtin_amdgcn_s_wait_tensorcnt(0);
#else
  for (int i = threadIdx.x; i < 4096; i += 256) biasLds[i] = bias[i];
#endif

  // Preload the whole K=256 worth of A fragments (8 x v16bf = 64 VGPRs).
  int arow = tm + (lane & 15);
  v16bf afr[8];
  #pragma unroll
  for (int kt = 0; kt < 8; ++kt) afr[kt] = load_a_rm(RC, arow, 256, kt * 32, lane);

  __syncthreads();   // bias in LDS visible to all waves; wpart zeroed

  int nbase = wave * 512;
  int half = lane >> 4, n = lane & 15;
  #pragma unroll 2
  for (int nt = 0; nt < 32; ++nt) {
    v8f acc = {0.f, 0.f, 0.f, 0.f, 0.f, 0.f, 0.f, 0.f};
    int ntile = (nbase >> 4) + nt;
    #pragma unroll
    for (int kt = 0; kt < 8; ++kt) {
      v16bf b = load_b_pk(W, kt, ntile, 256, lane);
      acc = wmma_bf16(afr[kt], b, acc);
    }
    int col = nbase + nt * 16 + n;
    float bb = biasLds[col];
    #pragma unroll
    for (int r = 0; r < 8; ++r)
      logits[(half * 8 + r) * 4096 + col] = acc[r] + bb;
  }
  __syncthreads();

  // Softmax + head-sum: wave w handles rows 2w, 2w+1.
  #pragma unroll
  for (int mi = 0; mi < 2; ++mi) {
    int m = wave * 2 + mi;
    const float* Lr = logits + m * 4096;
    float mx = -3.4e38f;
    for (int j = lane; j < 4096; j += 32) mx = fmaxf(mx, Lr[j]);
    for (int off = 16; off > 0; off >>= 1) mx = fmaxf(mx, __shfl_xor(mx, off, 32));
    float sm = 0.f;
    for (int j = lane; j < 4096; j += 32) sm += __expf(Lr[j] - mx);
    for (int off = 16; off > 0; off >>= 1) sm += __shfl_xor(sm, off, 32);
    float inv = 1.f / sm;
    for (int j = lane; j < 4096; j += 32)
      atomicAdd(&wpart[j & 511], __expf(Lr[j] - mx) * inv);
  }
  __syncthreads();
  for (int i = threadIdx.x; i < 512; i += 256)
    atomicAdd(&wacc[b_idx * 512 + i], wpart[i]);
}

// ---------------------------------------------------------------------------
// Head: rep_attn = sum_q (w[b,q]/H) * rep_conv[b,q,:]; c1 (leaky) ; c2 ; softmax
// One block per batch element.
// ---------------------------------------------------------------------------
__global__ __launch_bounds__(256) void head_kernel(
    const float* __restrict__ wacc, const u16* __restrict__ RC,
    const float* __restrict__ c1w, const float* __restrict__ c1b,
    const float* __restrict__ c2w, const float* __restrict__ c2b,
    float* __restrict__ out) {
  __shared__ float ra[256];
  __shared__ float hb[128];
  int b = blockIdx.x, t = threadIdx.x;
  float acc = 0.f;
  for (int q = 0; q < 512; ++q)
    acc += wacc[b * 512 + q] * bf2f(RC[((size_t)b * 512 + q) * 256 + t]);
  ra[t] = acc * 0.125f;  // 1/H
  __syncthreads();
  if (t < 128) {
    float hv = c1b[t];
    for (int p = 0; p < 256; ++p) hv += ra[p] * c1w[p * 128 + t];
    hb[t] = hv > 0.f ? hv : 0.01f * hv;  // leaky relu
  }
  __syncthreads();
  if (t == 0) {
    float a0 = c2b[0], a1 = c2b[1];
    for (int j = 0; j < 128; ++j) { a0 += hb[j] * c2w[j * 2]; a1 += hb[j] * c2w[j * 2 + 1]; }
    float mx = fmaxf(a0, a1);
    float e0 = __expf(a0 - mx), e1 = __expf(a1 - mx);
    float s = e0 + e1;
    out[b * 2 + 0] = e0 / s;
    out[b * 2 + 1] = e1 / s;
    out[64 + b]    = (e0 >= e1) ? 0.f : 1.f;  // argmax (first-max wins)
  }
}

// ---------------------------------------------------------------------------
extern "C" void kernel_launch(void* const* d_in, const int* in_sizes, int n_in,
                              void* d_out, int out_size, void* d_ws, size_t ws_size,
                              hipStream_t stream) {
  (void)in_sizes; (void)n_in; (void)out_size; (void)ws_size;
  const float* x      = (const float*)d_in[0];
  const int*   seqlen = (const int*)  d_in[1];
  const float* proj_w = (const float*)d_in[2];
  const float* proj_b = (const float*)d_in[3];
  const float* conv_w = (const float*)d_in[4];
  const float* conv_b = (const float*)d_in[5];
  const float* attn_w = (const float*)d_in[6];
  const float* attn_b = (const float*)d_in[7];
  const float* c1w    = (const float*)d_in[8];
  const float* c1b    = (const float*)d_in[9];
  const float* c2w    = (const float*)d_in[10];
  const float* c2b    = (const float*)d_in[11];
  float* out = (float*)d_out;

  char* ws = (char*)d_ws;
  size_t o = 0;
  u16* Xb = (u16*)(ws + o);  o += (size_t)16384 * 1024 * 2;  // 32 MB bf16 x
  u16* W1 = (u16*)(ws + o);  o += (size_t)1024 * 256 * 2;    // packed proj_w
  u16* W2 = (u16*)(ws + o);  o += (size_t)768 * 256 * 2;     // packed conv_w
  u16* W3 = (u16*)(ws + o);  o += (size_t)256 * 4096 * 2;    // packed attn_w
  u16* PR = (u16*)(ws + o);  o += (size_t)16384 * 256 * 2;   // proj bf16
  u16* RC = (u16*)(ws + o);  o += (size_t)16384 * 256 * 2;   // rep_conv bf16
  float* WA = (float*)(ws + o);                              // w accumulator (B*S)

  // Stage 0: conversions / packing / zeroing
  cvt_f32_bf16<<<16777216 / 256, 256, 0, stream>>>(x, Xb, 16777216);
  pack_w<<<262144 / 256, 256, 0, stream>>>(proj_w, W1, 0, 262144);
  pack_w<<<196608 / 256, 256, 0, stream>>>(conv_w, W2, 1, 196608);
  pack_w<<<1048576 / 256, 256, 0, stream>>>(attn_w, W3, 2, 1048576);
  zero_f32<<<64, 256, 0, stream>>>(WA, 16384);

  // Stage 1: proj GEMM (bf16 WMMA)
  gemm_proj_kernel<<<dim3(4, 256), 128, 0, stream>>>(Xb, W1, proj_b, PR);

  // Stage 2: conv-as-GEMM (bf16 WMMA) + relu + mask
  gemm_conv_kernel<<<dim3(4, 256), 128, 0, stream>>>(PR, W2, conv_b, seqlen, RC);

  // Stage 3: logits GEMM fused with softmax + head reduction (274KB dynamic LDS)
  const int shm = LDS_BIAS_OFF_B + 4096 * 4;
  hipFuncSetAttribute((const void*)attn_softmax_kernel,
                      hipFuncAttributeMaxDynamicSharedMemorySize, shm);
  attn_softmax_kernel<<<1024, 256, shm, stream>>>(RC, W3, attn_b, WA);

  // Stage 4: head
  head_kernel<<<32, 256, 0, stream>>>(WA, RC, c1w, c1b, c2w, c2b, out);
}
